// AttentionLayer_57904749084876
// MI455X (gfx1250) — matrix operand order repaired and checked by reading, not compiled, not run on hardware
//
#include <hip/hip_runtime.h>
#include <hip/hip_bf16.h>
#include <stdint.h>

// Problem constants (from reference)
#define B_ 8
#define N_ 1024
#define D_ 1024
#define H_ 16
#define DH_ 64
#define M_ (B_*N_)          // 8192 rows
#define EPS_ 1e-5f

typedef unsigned short bf16_t;
typedef unsigned int   u32;

typedef __attribute__((ext_vector_type(16))) __bf16 v16bf;
typedef __attribute__((ext_vector_type(8)))  float  v8f;
typedef __attribute__((ext_vector_type(8)))  u32    v8u;

union FragU { v8u u; v16bf b; };

__device__ __forceinline__ bf16_t f32_to_bf16(float f) {
  u32 x = __float_as_uint(f);
  u32 r = x + 0x7FFFu + ((x >> 16) & 1u);   // round-to-nearest-even
  return (bf16_t)(r >> 16);
}

__device__ __forceinline__ v8f zero8() {
  v8f z;
#pragma unroll
  for (int i = 0; i < 8; ++i) z[i] = 0.f;
  return z;
}

// CDNA5 async global->LDS copy (writes LDS directly, tracked by ASYNCcnt).
// Generic pointer to LDS: low 32 bits are the workgroup-relative LDS offset
// (flat LDS aperture keeps the offset in addr[31:0]).
__device__ __forceinline__ void async_b128(void* lds, const void* g) {
  u32 l = (u32)(uintptr_t)lds;
  asm volatile("global_load_async_to_lds_b128 %0, %1, off"
               :: "v"(l), "v"((unsigned long long)(uintptr_t)g)
               : "memory");
}
__device__ __forceinline__ void wait_async0() {
  asm volatile("s_wait_asynccnt 0x0" ::: "memory");
}

// XOR shuffle within the wave via ds_swizzle (group-of-32 mode).
// Masks <16 stay inside each 16-lane half, matching the WMMA C-matrix split.
template<int MASK>
__device__ __forceinline__ float swz_xor(float v) {
  int r = __builtin_amdgcn_ds_swizzle(__float_as_int(v), (MASK << 10) | 0x1F);
  return __int_as_float(r);
}

// ---------------------------------------------------------------------------
// Weight transpose + bf16 convert: W[K,N] f32 -> Wt[N,K] bf16
// ---------------------------------------------------------------------------
__global__ void k_cvt_wT(const float* __restrict__ W, bf16_t* __restrict__ Wt) {
  int idx = blockIdx.x * blockDim.x + threadIdx.x;
  if (idx >= D_ * D_) return;
  int n = idx / D_, k = idx % D_;
  Wt[idx] = f32_to_bf16(W[(size_t)k * D_ + n]);
}

// ---------------------------------------------------------------------------
// LayerNorm over D=1024, one block (256 thr) per row. OUT_BF16 selects dtype.
// ---------------------------------------------------------------------------
template<bool OUT_BF16>
__global__ __launch_bounds__(256)
void k_layernorm(const float* __restrict__ X, const float* __restrict__ g,
                 const float* __restrict__ b, void* __restrict__ outp) {
  __shared__ float red[256];
  int row = blockIdx.x;
  int t = threadIdx.x;
  const float* x = X + (size_t)row * D_;
  float v[4]; float s = 0.f;
#pragma unroll
  for (int i = 0; i < 4; ++i) { v[i] = x[t + 256 * i]; s += v[i]; }
  red[t] = s; __syncthreads();
  for (int off = 128; off > 0; off >>= 1) {
    if (t < off) red[t] += red[t + off];
    __syncthreads();
  }
  float mu = red[0] * (1.0f / D_);
  __syncthreads();
  float s2 = 0.f;
#pragma unroll
  for (int i = 0; i < 4; ++i) { float d = v[i] - mu; s2 += d * d; }
  red[t] = s2; __syncthreads();
  for (int off = 128; off > 0; off >>= 1) {
    if (t < off) red[t] += red[t + off];
    __syncthreads();
  }
  float rstd = rsqrtf(red[0] * (1.0f / D_) + EPS_);
#pragma unroll
  for (int i = 0; i < 4; ++i) {
    int c = t + 256 * i;
    float y = (v[i] - mu) * rstd * g[c] + b[c];
    if (OUT_BF16) ((bf16_t*)outp)[(size_t)row * D_ + c] = f32_to_bf16(y);
    else          ((float*)outp)[(size_t)row * D_ + c] = y;
  }
}

// ---------------------------------------------------------------------------
// GEMM: C[M,1024] = A[M,1024](bf16) @ Wt[1024,1024]^T (+bias), f32 out.
// 128x64 tile per block, 8 waves, each wave = 16 rows x 64 cols (4 WMMA accums).
// Double-buffered async global->LDS staging, K step 32: one barrier/iter.
// ---------------------------------------------------------------------------
#define GA_STR 40   // LDS row stride in halves (32 + pad), 80B rows (16B aligned)
#define BM_ 128
__global__ __launch_bounds__(256)
void k_gemm_bias(const bf16_t* __restrict__ A, const bf16_t* __restrict__ Wt,
                 const float* __restrict__ bias, float* __restrict__ C) {
  __shared__ bf16_t As[2][BM_ * GA_STR];
  __shared__ bf16_t Bs[2][64 * GA_STR];
  const int K = D_;
  int m0 = blockIdx.x * BM_;
  int n0 = blockIdx.y * 64;
  int t = threadIdx.x;
  int lane = t & 31, wave = t >> 5;     // wave = 16-row strip (0..7)
  int hi = lane >> 4, ln = lane & 15;

  // staging assignment (256 threads)
  int arow = t >> 1;           // 0..127
  int ak   = (t & 1) * 16;     // 0/16 halves
  int brow = t >> 2;           // 0..63
  int bk   = (t & 3) * 8;      // 0,8,16,24 halves

  const bf16_t* Ab = A  + (size_t)(m0 + arow) * K + ak;
  const bf16_t* Bb = Wt + (size_t)(n0 + brow) * K + bk;

  v8f acc[4];
#pragma unroll
  for (int c = 0; c < 4; ++c) acc[c] = zero8();

  // prologue: stage K-tile 0 into buffer 0
  async_b128(&As[0][arow * GA_STR + ak],     Ab);
  async_b128(&As[0][arow * GA_STR + ak + 8], Ab + 8);
  async_b128(&Bs[0][brow * GA_STR + bk],     Bb);

  for (int k0 = 0; k0 < K; k0 += 32) {
    int cur = (k0 >> 5) & 1;
    wait_async0();        // own async writes landed in LDS
    __syncthreads();      // everyone's writes visible; prior reads retired
    if (k0 + 32 < K) {
      int nxt = cur ^ 1;
      async_b128(&As[nxt][arow * GA_STR + ak],     Ab + k0 + 32);
      async_b128(&As[nxt][arow * GA_STR + ak + 8], Ab + k0 + 40);
      async_b128(&Bs[nxt][brow * GA_STR + bk],     Bb + k0 + 32);
    }

    FragU a;
#pragma unroll
    for (int j = 0; j < 8; ++j) {
      int ka = 2 * (j & 3) + ((j & 4) ? 16 : 0) + hi * 8;   // A-matrix layout
      a.u[j] = *(const u32*)(&As[cur][(wave * 16 + ln) * GA_STR + ka]);
    }
#pragma unroll
    for (int c = 0; c < 4; ++c) {
      FragU bf;
#pragma unroll
      for (int j = 0; j < 8; ++j) {
        int kb = 2 * (j & 3) + ((j & 4) ? 16 : 0) + hi * 8; // B-matrix layout
        bf.u[j] = *(const u32*)(&Bs[cur][(c * 16 + ln) * GA_STR + kb]);
      }
      acc[c] = __builtin_amdgcn_wmma_f32_16x16x32_bf16(false, a.b, false, bf.b,
                                                       (short)0, acc[c], false, false);
    }
  }

#pragma unroll
  for (int c = 0; c < 4; ++c) {
    int col = n0 + c * 16 + ln;
    float bb = bias ? bias[col] : 0.f;
#pragma unroll
    for (int j = 0; j < 8; ++j) {
      int row = m0 + wave * 16 + j + hi * 8;    // C layout: M = j + 8*hi
      C[(size_t)row * D_ + col] = acc[c][j] + bb;
    }
  }
}

// ---------------------------------------------------------------------------
// RoPE + pack to [B,H,N,DH] bf16 from projection tmp [B*N, D] f32
// ---------------------------------------------------------------------------
__global__ void k_rope_pack(const float* __restrict__ P, bf16_t* __restrict__ Obh) {
  int idx = blockIdx.x * blockDim.x + threadIdx.x;
  if (idx >= M_ * (D_ / 2)) return;
  int pair = idx % (D_ / 2);
  int row  = idx / (D_ / 2);
  int n = row % N_, b = row / N_;
  int h = pair / (DH_ / 2);
  int i = pair % (DH_ / 2);
  int d0 = h * DH_ + 2 * i;
  float x0 = P[(size_t)row * D_ + d0];
  float x1 = P[(size_t)row * D_ + d0 + 1];
  float f = expf(-((float)(2 * i) / (float)DH_) * 9.210340371976184f); // ln(1e4)
  float ang = (float)n * f;
  float c, s;
  sincosf(ang, &s, &c);
  float o0 = x0 * c - x1 * s;
  float o1 = x0 * s + x1 * c;
  size_t ob = (((size_t)b * H_ + h) * N_ + n) * DH_ + 2 * i;
  Obh[ob]     = f32_to_bf16(o0);
  Obh[ob + 1] = f32_to_bf16(o1);
}

// Pack V: tmp f32 [B*N, D] -> [B,H,N,DH] bf16
__global__ void k_pack_v(const float* __restrict__ P, bf16_t* __restrict__ Vbh) {
  int idx = blockIdx.x * blockDim.x + threadIdx.x;
  if (idx >= M_ * D_) return;
  int col = idx % D_, row = idx / D_;
  int n = row % N_, b = row / N_;
  int h = col / DH_, d = col % DH_;
  Vbh[(((size_t)b * H_ + h) * N_ + n) * DH_ + d] = f32_to_bf16(P[idx]);
}

// ---------------------------------------------------------------------------
// Flash attention: block = (b,h, 64-row q tile); 4 waves x 16 rows each.
// Q,K,V in [B,H,N,DH] bf16. O written to [B,N,D] bf16.
// Q/K tiles staged via async global->LDS; V transposed via vector scatter.
// ---------------------------------------------------------------------------
#define AT_STR 72   // LDS row stride (64 + pad) halves; 144B rows (16B aligned)
__global__ __launch_bounds__(128)
void k_attention(const bf16_t* __restrict__ Q, const bf16_t* __restrict__ Kmat,
                 const bf16_t* __restrict__ V, bf16_t* __restrict__ O) {
  __shared__ bf16_t Qs[64 * AT_STR];
  __shared__ bf16_t Ks[64 * AT_STR];
  __shared__ bf16_t Vt[64 * AT_STR];   // V transposed: [dh][key]
  __shared__ bf16_t Ps[64 * AT_STR];   // softmax probs, per-wave private rows

  int bh = blockIdx.y;
  int b = bh / H_, h = bh % H_;
  int q0 = blockIdx.x * 64;
  const size_t base = (size_t)bh * N_ * DH_;

  int t = threadIdx.x, lane = t & 31, wave = t >> 5;
  int hi = lane >> 4, ln = lane & 15;
  int myrow = wave * 16;

  int lrow = t >> 1;           // 0..63
  int lk   = (t & 1) * 32;     // 0 or 32

  { // stage Q tile once (async; first chunk's wait covers it)
    const bf16_t* Qg = Q + base + (size_t)q0 * DH_;
#pragma unroll
    for (int i = 0; i < 4; ++i)
      async_b128(&Qs[lrow * AT_STR + lk + 8 * i], &Qg[lrow * DH_ + lk + 8 * i]);
  }

  float mrun[8], lrun[8];
  v8f o[4];
#pragma unroll
  for (int j = 0; j < 8; ++j) { mrun[j] = -1e30f; lrun[j] = 0.f; }
#pragma unroll
  for (int d = 0; d < 4; ++d) o[d] = zero8();

  const float scale = 0.125f;  // DH^-0.5

  for (int kc = 0; kc < N_; kc += 64) {
    __syncthreads();           // prior reads of Ks/Vt retired before restage
    const bf16_t* Kg = Kmat + base + (size_t)kc * DH_;
    const bf16_t* Vg = V    + base + (size_t)kc * DH_;
#pragma unroll
    for (int i = 0; i < 4; ++i)
      async_b128(&Ks[lrow * AT_STR + lk + 8 * i], &Kg[lrow * DH_ + lk + 8 * i]);
#pragma unroll
    for (int i = 0; i < 4; ++i) {
      uint4 pk = *(const uint4*)(&Vg[lrow * DH_ + lk + 8 * i]);
      const bf16_t* pv = (const bf16_t*)&pk;
#pragma unroll
      for (int e = 0; e < 8; ++e)
        Vt[(lk + 8 * i + e) * AT_STR + lrow] = pv[e];
    }
    wait_async0();
    __syncthreads();

    // S = Q * K^T for this wave's 16 rows x 64 keys
    v8f s[4];
#pragma unroll
    for (int n4 = 0; n4 < 4; ++n4) s[n4] = zero8();
#pragma unroll
    for (int ks = 0; ks < 2; ++ks) {
      FragU a;
#pragma unroll
      for (int j = 0; j < 8; ++j) {
        int ka = 2 * (j & 3) + ((j & 4) ? 16 : 0) + hi * 8 + ks * 32;
        a.u[j] = *(const u32*)(&Qs[(myrow + ln) * AT_STR + ka]);
      }
#pragma unroll
      for (int n4 = 0; n4 < 4; ++n4) {
        FragU bf;
#pragma unroll
        for (int j = 0; j < 8; ++j) {
          int kb = 2 * (j & 3) + ((j & 4) ? 16 : 0) + hi * 8 + ks * 32;
          bf.u[j] = *(const u32*)(&Ks[(n4 * 16 + ln) * AT_STR + kb]);
        }
        s[n4] = __builtin_amdgcn_wmma_f32_16x16x32_bf16(false, a.b, false, bf.b,
                                                        (short)0, s[n4], false, false);
      }
    }

    // online softmax over the 64-key chunk (rows j + 8*hi)
    float alpha[8];
#pragma unroll
    for (int j = 0; j < 8; ++j) {
      float mx = -1e30f;
#pragma unroll
      for (int n4 = 0; n4 < 4; ++n4) { s[n4][j] *= scale; mx = fmaxf(mx, s[n4][j]); }
      mx = fmaxf(mx, swz_xor<1>(mx));
      mx = fmaxf(mx, swz_xor<2>(mx));
      mx = fmaxf(mx, swz_xor<4>(mx));
      mx = fmaxf(mx, swz_xor<8>(mx));
      float mnew = fmaxf(mrun[j], mx);
      alpha[j] = __expf(mrun[j] - mnew);
      float rs = 0.f;
#pragma unroll
      for (int n4 = 0; n4 < 4; ++n4) {
        float p = __expf(s[n4][j] - mnew);
        s[n4][j] = p;
        rs += p;
      }
      rs += swz_xor<1>(rs);
      rs += swz_xor<2>(rs);
      rs += swz_xor<4>(rs);
      rs += swz_xor<8>(rs);
      lrun[j] = lrun[j] * alpha[j] + rs;
      mrun[j] = mnew;
    }

    // write P (own rows only -> same-wave LDS, in-order, no barrier needed)
#pragma unroll
    for (int n4 = 0; n4 < 4; ++n4)
#pragma unroll
      for (int j = 0; j < 8; ++j)
        Ps[(myrow + j + hi * 8) * AT_STR + n4 * 16 + ln] = f32_to_bf16(s[n4][j]);

#pragma unroll
    for (int d = 0; d < 4; ++d)
#pragma unroll
      for (int j = 0; j < 8; ++j)
        o[d][j] *= alpha[j];

    // O += P @ V_chunk
#pragma unroll
    for (int ks = 0; ks < 2; ++ks) {
      FragU a;
#pragma unroll
      for (int j = 0; j < 8; ++j) {
        int ka = 2 * (j & 3) + ((j & 4) ? 16 : 0) + hi * 8 + ks * 32;
        a.u[j] = *(const u32*)(&Ps[(myrow + ln) * AT_STR + ka]);
      }
#pragma unroll
      for (int d = 0; d < 4; ++d) {
        FragU bf;
#pragma unroll
        for (int j = 0; j < 8; ++j) {
          int kb = 2 * (j & 3) + ((j & 4) ? 16 : 0) + hi * 8 + ks * 32;
          bf.u[j] = *(const u32*)(&Vt[(d * 16 + ln) * AT_STR + kb]);
        }
        o[d] = __builtin_amdgcn_wmma_f32_16x16x32_bf16(false, a.b, false, bf.b,
                                                       (short)0, o[d], false, false);
      }
    }
  }

  // finalize: O /= l, scatter into [B,N,D] with head offset
#pragma unroll
  for (int j = 0; j < 8; ++j) {
    float inv = 1.0f / lrun[j];
    int row = q0 + myrow + j + hi * 8;
#pragma unroll
    for (int d = 0; d < 4; ++d) {
      int col = h * DH_ + d * 16 + ln;
      O[((size_t)b * N_ + row) * D_ + col] = f32_to_bf16(o[d][j] * inv);
    }
  }
}

// ---------------------------------------------------------------------------
extern "C" void kernel_launch(void* const* d_in, const int* in_sizes, int n_in,
                              void* d_out, int out_size, void* d_ws, size_t ws_size,
                              hipStream_t stream) {
  (void)in_sizes; (void)n_in; (void)out_size; (void)ws_size;
  const float* x    = (const float*)d_in[0];
  const float* src  = (const float*)d_in[1];
  const float* nq_g = (const float*)d_in[2];
  const float* nq_b = (const float*)d_in[3];
  const float* nk_g = (const float*)d_in[4];
  const float* nk_b = (const float*)d_in[5];
  const float* nv_g = (const float*)d_in[6];
  const float* nv_b = (const float*)d_in[7];
  const float* Wq   = (const float*)d_in[8];
  const float* bq   = (const float*)d_in[9];
  const float* Wk   = (const float*)d_in[10];
  const float* bk   = (const float*)d_in[11];
  const float* Wv   = (const float*)d_in[12];
  const float* bv   = (const float*)d_in[13];
  const float* Wo   = (const float*)d_in[14];
  const float* lng  = (const float*)d_in[15];
  const float* lnb  = (const float*)d_in[16];
  float* out = (float*)d_out;

  char* ws = (char*)d_ws;
  size_t off = 0;
  auto alloc = [&](size_t bytes) -> void* {
    void* p = ws + off;
    off += (bytes + 255) & ~(size_t)255;
    return p;
  };
  float*  tmp  = (float*) alloc((size_t)M_ * D_ * 4);   // 32 MB, reused
  bf16_t* qln  = (bf16_t*)alloc((size_t)M_ * D_ * 2);
  bf16_t* kln  = (bf16_t*)alloc((size_t)M_ * D_ * 2);
  bf16_t* vln  = (bf16_t*)alloc((size_t)M_ * D_ * 2);
  bf16_t* WqT  = (bf16_t*)alloc((size_t)D_ * D_ * 2);
  bf16_t* WkT  = (bf16_t*)alloc((size_t)D_ * D_ * 2);
  bf16_t* WvT  = (bf16_t*)alloc((size_t)D_ * D_ * 2);
  bf16_t* WoT  = (bf16_t*)alloc((size_t)D_ * D_ * 2);
  bf16_t* obuf = (bf16_t*)alloc((size_t)M_ * D_ * 2);
  // LN buffers are fully consumed by their GEMM before repacking -> alias them
  bf16_t* qbh = qln;
  bf16_t* kbh = kln;
  bf16_t* vbh = vln;

  int wblocks = (D_ * D_ + 255) / 256;
  k_cvt_wT<<<wblocks, 256, 0, stream>>>(Wq, WqT);
  k_cvt_wT<<<wblocks, 256, 0, stream>>>(Wk, WkT);
  k_cvt_wT<<<wblocks, 256, 0, stream>>>(Wv, WvT);
  k_cvt_wT<<<wblocks, 256, 0, stream>>>(Wo, WoT);

  k_layernorm<true><<<M_, 256, 0, stream>>>(src, nq_g, nq_b, qln);
  k_layernorm<true><<<M_, 256, 0, stream>>>(x,   nk_g, nk_b, kln);
  k_layernorm<true><<<M_, 256, 0, stream>>>(x,   nv_g, nv_b, vln);

  dim3 gg(M_ / BM_, D_ / 64);
  int rblocks = (M_ * (D_ / 2) + 255) / 256;
  int pblocks = (M_ * D_ + 255) / 256;

  k_gemm_bias<<<gg, 256, 0, stream>>>(qln, WqT, bq, tmp);
  k_rope_pack<<<rblocks, 256, 0, stream>>>(tmp, qbh);

  k_gemm_bias<<<gg, 256, 0, stream>>>(kln, WkT, bk, tmp);
  k_rope_pack<<<rblocks, 256, 0, stream>>>(tmp, kbh);

  k_gemm_bias<<<gg, 256, 0, stream>>>(vln, WvT, bv, tmp);
  k_pack_v<<<pblocks, 256, 0, stream>>>(tmp, vbh);

  dim3 ga(N_ / 64, B_ * H_);
  k_attention<<<ga, 128, 0, stream>>>(qbh, kbh, vbh, obuf);

  k_gemm_bias<<<gg, 256, 0, stream>>>(obuf, WoT, nullptr, tmp);
  k_layernorm<false><<<M_, 256, 0, stream>>>(tmp, lng, lnb, out);
}
